// MultiHeadAttentionLayer_39505109188985
// MI455X (gfx1250) — matrix operand order repaired and checked by reading, not compiled, not run on hardware
//
#include <hip/hip_runtime.h>
#include <hip/hip_fp16.h>

// ---------------------------------------------------------------------------
// MHA layer for gfx1250 (MI455X): QKV GEMM + flash attention.
//  - all matrix math via v_wmma_f32_16x16x32_{bf16,f16} (wave32 layouts)
//  - attention K/V tiles staged into LDS double-buffers by the Tensor Data
//    Mover (tensor_load_to_lds + s_wait_tensorcnt), overlapping DMA with WMMA
//  - global_prefetch for next GEMM tiles
// ---------------------------------------------------------------------------

typedef __attribute__((ext_vector_type(16))) __bf16    v16bf;
typedef __attribute__((ext_vector_type(8)))  __bf16    v8bf;
typedef __attribute__((ext_vector_type(16))) _Float16  v16h;
typedef __attribute__((ext_vector_type(8)))  _Float16  v8h;
typedef __attribute__((ext_vector_type(8)))  float     v8f;

union Frag16BF { v16bf v; v8bf h[2]; };
union Frag16H  { v16h  v; v8h  h[2]; };

#define BATCH 2
#define SEQ   2048
#define DIM   1024
#define HEADS 16
#define HDIM  64
#define N3    3072          // 3*DIM
#define MTOT  (BATCH*SEQ)   // 4096

// ---- Tensor Data Mover availability (device pass only) --------------------
#if defined(__AMDGCN__) && defined(__has_builtin)
#if __has_builtin(__builtin_amdgcn_tensor_load_to_lds) && \
    __has_builtin(__builtin_amdgcn_s_wait_tensorcnt)
#define HAVE_TDM 1
#endif
#endif
#ifndef HAVE_TDM
#define HAVE_TDM 0
#endif

#if HAVE_TDM
typedef __attribute__((ext_vector_type(4))) unsigned int u32x4;
typedef __attribute__((ext_vector_type(4))) int          i32x4;
typedef __attribute__((ext_vector_type(8))) int          i32x8;

// LDS byte offset of a pointer into shared memory (flat -> AS3).
__device__ __forceinline__ unsigned lds_off(const void* p) {
  return (unsigned)(unsigned long long)
      (__attribute__((address_space(3))) const char*)(const char*)p;
}

// Issue a 2D TDM tile load: `rows` rows of `dim0` 2-byte elements, row stride
// `stride0` (elements) in global memory, packed contiguously into LDS.
// D# per CDNA5 ISA 8.3/8.4 (group0: lds/global addr, type=2; group1: dims).
__device__ __forceinline__ void tdm_load_2d(const void* gptr, unsigned lds_byte,
                                            unsigned dim0, unsigned rows,
                                            unsigned long long stride0) {
  unsigned long long ga = (unsigned long long)gptr;
  u32x4 g0;
  g0[0] = 1u;                                           // count=1, user D#
  g0[1] = lds_byte;                                     // lds_addr
  g0[2] = (unsigned)(ga & 0xFFFFFFFFu);                 // global_addr[31:0]
  g0[3] = (unsigned)((ga >> 32) & 0x1FFFFFFu) | (2u << 30);  // [56:32]|type=2
  i32x8 g1;
  g1[0] = (int)(1u << 16);                              // data_size=2B, mask=0
  g1[1] = (int)((dim0 & 0xFFFFu) << 16);                // tensor_dim0[15:0]
  g1[2] = (int)((dim0 >> 16) | ((rows & 0xFFFFu) << 16));   // dim0 hi|dim1 lo
  g1[3] = (int)((rows >> 16) | ((dim0 & 0xFFFFu) << 16));   // dim1 hi|tile_dim0
  g1[4] = (int)(rows & 0xFFFFu);                        // tile_dim1, tile_dim2=0
  g1[5] = (int)(stride0 & 0xFFFFFFFFull);               // tensor_dim0_stride lo
  g1[6] = (int)((stride0 >> 32) & 0xFFFFull);           // stride hi, stride1=0
  g1[7] = 0;
  i32x4 z4 = {};
#if __clang_major__ >= 23
  i32x8 z8 = {};
  __builtin_amdgcn_tensor_load_to_lds(g0, g1, z4, z4, z8, 0);
#else
  __builtin_amdgcn_tensor_load_to_lds(g0, g1, z4, z4, 0);
#endif
}
#endif  // HAVE_TDM

// ===========================================================================
// Kernel 1: qkv = images @ W_qkv, fp32 in, bf16/f16 out in attention layouts.
//   Q,K -> bf16 [B,H,S,Dh] ; V -> f16 [B,H,Dh,S] (transposed)
// Block: 256 threads (8 waves), tile 128(M) x 128(N), K-step 32.
// ===========================================================================
__global__ __launch_bounds__(256)
void qkv_gemm_wmma(const float* __restrict__ img, const float* __restrict__ W,
                   __bf16* __restrict__ Qb, __bf16* __restrict__ Kb,
                   _Float16* __restrict__ Vt) {
  __shared__ __align__(64) __bf16 As[128][32];   // A tile, row-major
  __shared__ __align__(64) __bf16 Bst[128][32];  // B tile, transposed [n][k]

  const int tid  = threadIdx.x;
  const int lane = tid & 31;
  const int wave = tid >> 5;
  const int r    = lane & 15;     // row-in-half / column lane
  const int hf   = lane >> 4;     // which 16-lane half
  const int wm   = wave >> 1;     // wave M index 0..3 (32 rows each)
  const int wn   = wave & 1;      // wave N index 0..1 (64 cols each)
  const int m0   = blockIdx.x * 128;
  const int n0   = blockIdx.y * 128;

  v8f acc[2][4] = {};             // 2 M-subtiles x 4 N-subtiles of 16x16 f32

  const int arow = tid >> 1, ack = (tid & 1) * 16;   // A staging: 16 elems/thr
  const int bk   = tid >> 3, bcn = (tid & 7) * 16;   // B staging: 16 elems/thr

  for (int kk = 0; kk < DIM; kk += 32) {
    // prefetch next K-step tiles into cache (global_prefetch_b8)
    if (kk + 32 < DIM) {
      __builtin_prefetch(img + (size_t)(m0 + arow) * DIM + kk + 32 + ack, 0, 0);
      __builtin_prefetch(W + (size_t)(kk + 32 + bk) * N3 + n0 + bcn, 0, 0);
    }
    // ---- stage A (convert fp32 -> bf16), contiguous 64B global reads ----
    {
      const float4* sp = reinterpret_cast<const float4*>(
          img + (size_t)(m0 + arow) * DIM + kk + ack);
      Frag16BF fa;
      #pragma unroll
      for (int q = 0; q < 4; ++q) {
        float4 t = sp[q];
        fa.v[4*q + 0] = (__bf16)t.x;
        fa.v[4*q + 1] = (__bf16)t.y;
        fa.v[4*q + 2] = (__bf16)t.z;
        fa.v[4*q + 3] = (__bf16)t.w;
      }
      *reinterpret_cast<v16bf*>(&As[arow][ack]) = fa.v;
    }
    // ---- stage B transposed (convert fp32 -> bf16) ----
    {
      const float4* sp = reinterpret_cast<const float4*>(
          W + (size_t)(kk + bk) * N3 + n0 + bcn);
      #pragma unroll
      for (int q = 0; q < 4; ++q) {
        float4 t = sp[q];
        Bst[bcn + 4*q + 0][bk] = (__bf16)t.x;
        Bst[bcn + 4*q + 1][bk] = (__bf16)t.y;
        Bst[bcn + 4*q + 2][bk] = (__bf16)t.z;
        Bst[bcn + 4*q + 3][bk] = (__bf16)t.w;
      }
    }
    __syncthreads();

    // ---- load fragments (A 16x32 / B 32x16, both in contiguous k = 16*half + s order) ----
    v16bf af[2];
    #pragma unroll
    for (int ms = 0; ms < 2; ++ms) {
      const __bf16* p = &As[wm*32 + ms*16 + r][hf * 16];
      Frag16BF f;
      f.h[0] = *reinterpret_cast<const v8bf*>(p);
      f.h[1] = *reinterpret_cast<const v8bf*>(p + 8);
      af[ms] = f.v;
    }
    v16bf bfr[4];
    #pragma unroll
    for (int ns = 0; ns < 4; ++ns)
      bfr[ns] = *reinterpret_cast<const v16bf*>(&Bst[wn*64 + ns*16 + r][hf * 16]);

    #pragma unroll
    for (int ms = 0; ms < 2; ++ms)
      #pragma unroll
      for (int ns = 0; ns < 4; ++ns)
        acc[ms][ns] = __builtin_amdgcn_wmma_f32_16x16x32_bf16(
            false, af[ms], false, bfr[ns], (short)0, acc[ms][ns], false, false);
    __syncthreads();
  }

  // ---- epilogue: scatter into Q / K / V^T layouts ----
  #pragma unroll
  for (int ms = 0; ms < 2; ++ms)
    #pragma unroll
    for (int ns = 0; ns < 4; ++ns)
      #pragma unroll
      for (int v = 0; v < 8; ++v) {
        const int m = m0 + wm*32 + ms*16 + v + hf*8;   // C row = v + 8*half
        const int n = n0 + wn*64 + ns*16 + r;          // C col = lane&15
        const float val = acc[ms][ns][v];
        const int b = m >> 11, s = m & (SEQ - 1);
        const int region = n >> 10, d = n & (DIM - 1);
        const int h = d >> 6, dh = d & (HDIM - 1);
        const size_t bh = (size_t)(b * HEADS + h);
        if (region == 0)
          Qb[(bh * SEQ + s) * HDIM + dh] = (__bf16)val;
        else if (region == 1)
          Kb[(bh * SEQ + s) * HDIM + dh] = (__bf16)val;
        else
          Vt[(bh * HDIM + dh) * SEQ + s] = (_Float16)val;
      }
}

// ===========================================================================
// Kernel 2: flash attention. One block = (b,h,64 q-rows); 4 waves, each wave
// owns 16 q rows. 32-key steps: 4 bf16 WMMAs (scores) + online softmax with
// 16-lane shfl_xor reductions + 4 f16 WMMAs (P @ V). K/V tiles are staged
// into LDS double buffers by the TDM (wave 0 issues, s_wait_tensorcnt +
// workgroup barrier publishes), overlapping DMA with WMMA compute.
// ===========================================================================
__global__ __launch_bounds__(128)
void flash_attn_wmma(const __bf16* __restrict__ Qb, const __bf16* __restrict__ Kb,
                     const _Float16* __restrict__ Vt, float* __restrict__ out) {
  __shared__ __align__(64) _Float16 Pl[4][16][32];    // per-wave probs tile
#if HAVE_TDM
  __shared__ __align__(128) __bf16   KtS[2][32][64];  // K tile double buffer
  __shared__ __align__(128) _Float16 VtS[2][64][32];  // V^T tile double buffer
#endif

  const int tid  = threadIdx.x;
  const int lane = tid & 31;
  const int w    = tid >> 5;
  const int r    = lane & 15;
  const int hf   = lane >> 4;

  const int bh   = blockIdx.x >> 5;          // b*HEADS + h
  const int qblk = blockIdx.x & 31;
  const int b    = bh >> 4, h = bh & 15;
  const int q0   = qblk * 64 + w * 16;
  const int qrow = q0 + r;

  const float scale = 0.125f;                 // 1/sqrt(64)
  const float L2E   = 1.4426950408889634f;

  // Q fragments: A-matrix 16x32, two K-chunks covering Dh=64
  Frag16BF qf[2];
  #pragma unroll
  for (int c = 0; c < 2; ++c) {
    const __bf16* qp = Qb + ((size_t)bh * SEQ + qrow) * HDIM + c*32 + hf*16;
    qf[c].h[0] = *reinterpret_cast<const v8bf*>(qp);
    qf[c].h[1] = *reinterpret_cast<const v8bf*>(qp + 8);
  }

  float mrun[8], lrun[8];
  #pragma unroll
  for (int v = 0; v < 8; ++v) { mrun[v] = -1e30f; lrun[v] = 0.0f; }
  v8f acco[4] = {};                           // out accum, 4 tiles over Dh

#if HAVE_TDM
  // prologue: TDM-stage first K/V tiles (kv = 0) into buffer 0
  if (tid < 32) {
    tdm_load_2d(Kb + (size_t)bh * SEQ * HDIM, lds_off(&KtS[0][0][0]),
                HDIM, 32, HDIM);
    tdm_load_2d(Vt + (size_t)bh * HDIM * SEQ, lds_off(&VtS[0][0][0]),
                32, HDIM, SEQ);
    __builtin_amdgcn_s_wait_tensorcnt(0);
  }
  __syncthreads();
#endif

  for (int kv = 0; kv < SEQ; kv += 32) {
    const int cur = (kv >> 5) & 1;
#if HAVE_TDM
    // kick off DMA of the next K/V tiles into the other buffer
    if (kv + 32 < SEQ && tid < 32) {
      const int nxt = cur ^ 1;
      tdm_load_2d(Kb + ((size_t)bh * SEQ + kv + 32) * HDIM,
                  lds_off(&KtS[nxt][0][0]), HDIM, 32, HDIM);
      tdm_load_2d(Vt + (size_t)bh * HDIM * SEQ + kv + 32,
                  lds_off(&VtS[nxt][0][0]), 32, HDIM, SEQ);
    }
#endif
    // ---- scores: Q(16x64) @ K^T(64x32) -> two 16x16 tiles ----
    v8f sacc[2] = {};
    #pragma unroll
    for (int t = 0; t < 2; ++t)
      #pragma unroll
      for (int c = 0; c < 2; ++c) {
#if HAVE_TDM
        const __bf16* kp = &KtS[cur][t*16 + r][c*32 + hf*16];
#else
        const __bf16* kp =
            Kb + ((size_t)bh * SEQ + (kv + t*16 + r)) * HDIM + c*32 + hf*16;
#endif
        v16bf kfv = *reinterpret_cast<const v16bf*>(kp);
        sacc[t] = __builtin_amdgcn_wmma_f32_16x16x32_bf16(
            false, qf[c].v, false, kfv, (short)0, sacc[t], false, false);
      }

    // ---- online softmax (row = v + 8*half, 16 N-values per lane-half) ----
    #pragma unroll
    for (int v = 0; v < 8; ++v) {
      const float x0 = sacc[0][v] * scale;
      const float x1 = sacc[1][v] * scale;
      float ml = fmaxf(x0, x1);
      #pragma unroll
      for (int off = 1; off < 16; off <<= 1)
        ml = fmaxf(ml, __shfl_xor(ml, off, 32));
      const float mnew = fmaxf(mrun[v], ml);
      const float fac  = exp2f((mrun[v] - mnew) * L2E);
      mrun[v] = mnew;
      const float p0 = exp2f((x0 - mnew) * L2E);
      const float p1 = exp2f((x1 - mnew) * L2E);
      float rs = p0 + p1;
      #pragma unroll
      for (int off = 1; off < 16; off <<= 1)
        rs += __shfl_xor(rs, off, 32);
      lrun[v] = lrun[v] * fac + rs;
      #pragma unroll
      for (int t = 0; t < 4; ++t) acco[t][v] *= fac;
      Pl[w][v + hf*8][r]      = (_Float16)p0;   // re-shape C-layout -> memory
      Pl[w][v + hf*8][16 + r] = (_Float16)p1;
    }
    __syncthreads();

    // ---- P(16x32,f16) @ V(32x64) ----
    Frag16H pf;
    const _Float16* pp = &Pl[w][r][hf * 16];
    pf.h[0] = *reinterpret_cast<const v8h*>(pp);
    pf.h[1] = *reinterpret_cast<const v8h*>(pp + 8);
    #pragma unroll
    for (int d = 0; d < 4; ++d) {
#if HAVE_TDM
      const _Float16* vp = &VtS[cur][d*16 + r][hf * 16];
#else
      const _Float16* vp =
          Vt + ((size_t)bh * HDIM + d*16 + r) * SEQ + kv + hf*16;
#endif
      v16h vfv = *reinterpret_cast<const v16h*>(vp);
      acco[d] = __builtin_amdgcn_wmma_f32_16x16x32_f16(
          false, pf.v, false, vfv, (short)0, acco[d], false, false);
    }
#if HAVE_TDM
    if (tid < 32) __builtin_amdgcn_s_wait_tensorcnt(0);  // next tiles landed
#endif
    __syncthreads();
  }

  // ---- normalize and store fp32 out[b][s][h*64 + d] ----
  #pragma unroll
  for (int d = 0; d < 4; ++d)
    #pragma unroll
    for (int v = 0; v < 8; ++v) {
      const int row = v + hf*8;
      const float val = acco[d][v] / lrun[v];
      out[((size_t)b * SEQ + q0 + row) * DIM + h * HDIM + d*16 + r] = val;
    }
}

// ===========================================================================
extern "C" void kernel_launch(void* const* d_in, const int* in_sizes, int n_in,
                              void* d_out, int out_size, void* d_ws, size_t ws_size,
                              hipStream_t stream) {
  const float* img = (const float*)d_in[0];   // [2,2048,1024] fp32
  const float* W   = (const float*)d_in[1];   // [1024,3072]   fp32
  float* out = (float*)d_out;                 // [2,2048,1024] fp32

  const size_t NQK = (size_t)BATCH * HEADS * SEQ * HDIM;  // 4,194,304
  __bf16*   Qb = (__bf16*)d_ws;
  __bf16*   Kb = Qb + NQK;
  _Float16* Vt = (_Float16*)(Kb + NQK);       // 24 MB total workspace

  dim3 g1(MTOT / 128, N3 / 128);              // (32, 24)
  qkv_gemm_wmma<<<g1, 256, 0, stream>>>(img, W, Qb, Kb, Vt);

  const int nblk = BATCH * HEADS * (SEQ / 64);  // 1024
  flash_attn_wmma<<<nblk, 128, 0, stream>>>(Qb, Kb, Vt, out);
}
